// MambaTSAD_88519275970824
// MI455X (gfx1250) — compile-verified, matrix operand order checked by reading, not run on hardware
//
#include <hip/hip_runtime.h>
#include <hip/hip_bf16.h>
#include <cstdint>

// ---------------------------------------------------------------------------
// Types for WMMA fragments (gfx1250, wave32)
// ---------------------------------------------------------------------------
typedef __attribute__((ext_vector_type(16))) _Float16 v16h;
typedef __attribute__((ext_vector_type(8)))  _Float16 v8h;
typedef __attribute__((ext_vector_type(8)))  float    v8f;

#define WAVE 32
#define TILE_M 128
#define TILE_N 128
#define TILE_K 32
#define LDS_STRIDE 40   // halves per K-slab row (80 B: 16B aligned, bank-skewed)

#define ACT_NONE     0
#define ACT_GELU     2
#define ACT_SOFTPLUS 3

// ---------------------------------------------------------------------------
// Async global->LDS copy (CDNA5 GLOBAL_LOAD_ASYNC_TO_LDS_B128, ASYNCcnt).
// ROCm clang prototype (from diagnostics): first arg is a GENERIC
// `int __attribute__((vector_size(16)))*` (global source); LDS dest is the
// AS(3) variant. Guarded by __has_builtin with fallback staging.
// ---------------------------------------------------------------------------
#if __has_builtin(__builtin_amdgcn_global_load_async_to_lds_b128)
#define HAVE_ASYNC_LDS 1
typedef int b128_t __attribute__((vector_size(16)));
typedef __attribute__((address_space(3))) b128_t b128_lds_t;
static __device__ inline void async_cp16(void* lds_dst, const void* gsrc) {
    // AS(3) offset == low 32 bits of the flat shared address (aperture rule)
    __builtin_amdgcn_global_load_async_to_lds_b128(
        (b128_t*)gsrc,
        (b128_lds_t*)(uintptr_t)(uint32_t)(uintptr_t)lds_dst,
        0, 0);
}
static __device__ inline void wait_async() {
#if __has_builtin(__builtin_amdgcn_s_wait_asynccnt)
    __builtin_amdgcn_s_wait_asynccnt(0);
#else
    asm volatile("s_wait_asynccnt 0x0" ::: "memory");
#endif
}
#else
#define HAVE_ASYNC_LDS 0
#endif

static __device__ inline v16h mk16(v8h lo, v8h hi) {
    v16h r;
#pragma unroll
    for (int i = 0; i < 8; ++i) { r[i] = lo[i]; r[i + 8] = hi[i]; }
    return r;
}

// ---------------------------------------------------------------------------
// f32 -> f16 convert (weights)
// ---------------------------------------------------------------------------
__global__ __launch_bounds__(256) void cvt_f32_f16(const float* __restrict__ in,
                                                   _Float16* __restrict__ out, int n) {
    int i = blockIdx.x * 256 + threadIdx.x;
    if (i < n) out[i] = (_Float16)in[i];
}

// ---------------------------------------------------------------------------
// LayerNorm: one block per row of D floats; writes f16 for WMMA consumption
// ---------------------------------------------------------------------------
__global__ __launch_bounds__(256) void ln_kernel(const float* __restrict__ x,
                                                 const float* __restrict__ g,
                                                 const float* __restrict__ b,
                                                 _Float16* __restrict__ out, int D) {
    __shared__ float red[512];
    const int tid = threadIdx.x;
    const size_t row = blockIdx.x;
    const float* xr = x + row * D;
    float s = 0.f, ss = 0.f;
    for (int i = tid; i < D; i += 256) { float v = xr[i]; s += v; ss += v * v; }
    red[tid] = s; red[256 + tid] = ss;
    __syncthreads();
    for (int st = 128; st > 0; st >>= 1) {
        if (tid < st) { red[tid] += red[tid + st]; red[256 + tid] += red[256 + tid + st]; }
        __syncthreads();
    }
    const float mu  = red[0] / (float)D;
    const float var = red[256] / (float)D - mu * mu;
    const float rs  = rsqrtf(var + 1e-5f);
    for (int i = tid; i < D; i += 256)
        out[row * D + i] = (_Float16)((xr[i] - mu) * rs * g[i] + b[i]);
}

// ---------------------------------------------------------------------------
// Generic WMMA GEMM: C[M,N] = act(A[M,K] * W[N,K]^T + bias) (+ residual)
// A, W are f16 row-major. Outputs optionally f32 and/or f16.
// Block: 256 threads = 8 waves; block tile 128x128; wave tile 32x64.
// K slabs staged to LDS via async copies (ASYNCcnt) when available.
// ---------------------------------------------------------------------------
__global__ __launch_bounds__(256) void gemm_wmma_f16(
    const _Float16* __restrict__ A, const _Float16* __restrict__ W,
    const float* __restrict__ bias, const float* __restrict__ residual,
    float* __restrict__ outF, _Float16* __restrict__ outH,
    int M, int N, int K, int act) {

    __shared__ __align__(16) _Float16 As[TILE_M * LDS_STRIDE];
    __shared__ __align__(16) _Float16 Bs[TILE_N * LDS_STRIDE];

    const int tid  = threadIdx.x;
    const int lane = tid & (WAVE - 1);
    const int wid  = tid >> 5;     // 0..7
    const int wm   = wid >> 1;     // 0..3 : M offset 32*wm
    const int wn   = wid & 1;      // 0..1 : N offset 64*wn
    const int m0   = blockIdx.y * TILE_M;
    const int n0   = blockIdx.x * TILE_N;

    v8f acc[2][4];
#pragma unroll
    for (int i = 0; i < 2; ++i)
#pragma unroll
        for (int j = 0; j < 4; ++j)
#pragma unroll
            for (int r = 0; r < 8; ++r) acc[i][j][r] = 0.f;

    const int fm = lane & 15;                 // M (or N) within 16
    const int kb = (lane < 16) ? 0 : 8;       // K base per documented layout

    const int r    = tid >> 1;                // 0..127  (tile row loaded by this thread)
    const int off  = (tid & 1) * 16;          // half-offset within 32-wide K slab

    for (int k0 = 0; k0 < K; k0 += TILE_K) {
#if HAVE_ASYNC_LDS
        // ---- async-stage A and B slabs [128 x 32] straight into LDS ----
        {
            const _Float16* srcA = A + (size_t)(m0 + r) * K + k0 + off;
            async_cp16(&As[r * LDS_STRIDE + off],     srcA);
            async_cp16(&As[r * LDS_STRIDE + off + 8], srcA + 8);
            if (k0 + TILE_K < K)
                __builtin_prefetch(srcA + TILE_K, 0, 1);     // global_prefetch_b8
            const int n = n0 + r;
            if (n < N) {
                const _Float16* srcB = W + (size_t)n * K + k0 + off;
                async_cp16(&Bs[r * LDS_STRIDE + off],     srcB);
                async_cp16(&Bs[r * LDS_STRIDE + off + 8], srcB + 8);
                if (k0 + TILE_K < K)
                    __builtin_prefetch(srcB + TILE_K, 0, 1);
            } else {
                v8h z;
#pragma unroll
                for (int i = 0; i < 8; ++i) z[i] = (_Float16)0.f;
                *(v8h*)&Bs[r * LDS_STRIDE + off]     = z;
                *(v8h*)&Bs[r * LDS_STRIDE + off + 8] = z;
            }
        }
        wait_async();          // s_wait_asynccnt 0 (own copies done)
        __syncthreads();       // all waves' copies visible
#else
        // ---- fallback: global->VGPR->ds_store staging ----
        {
            const _Float16* src = A + (size_t)(m0 + r) * K + k0 + off;
            v8h a0 = *(const v8h*)(src);
            v8h a1 = *(const v8h*)(src + 8);
            *(v8h*)&As[r * LDS_STRIDE + off]     = a0;
            *(v8h*)&As[r * LDS_STRIDE + off + 8] = a1;
            if (k0 + TILE_K < K)
                __builtin_prefetch(src + TILE_K, 0, 1);
        }
        {
            v8h b0, b1;
#pragma unroll
            for (int i = 0; i < 8; ++i) { b0[i] = (_Float16)0.f; b1[i] = (_Float16)0.f; }
            const int n = n0 + r;
            if (n < N) {
                const _Float16* src = W + (size_t)n * K + k0 + off;
                b0 = *(const v8h*)(src);
                b1 = *(const v8h*)(src + 8);
                if (k0 + TILE_K < K)
                    __builtin_prefetch(src + TILE_K, 0, 1);
            }
            *(v8h*)&Bs[r * LDS_STRIDE + off]     = b0;
            *(v8h*)&Bs[r * LDS_STRIDE + off + 8] = b1;
        }
        __syncthreads();
#endif

        // ---- build fragments (two ds_load_b128 each, per ISA 16-bit layout) ----
        v16h afrag[2];
#pragma unroll
        for (int i = 0; i < 2; ++i) {
            const int m = wm * 32 + i * 16 + fm;
            v8h lo = *(const v8h*)&As[m * LDS_STRIDE + kb];
            v8h hi = *(const v8h*)&As[m * LDS_STRIDE + kb + 16];
            afrag[i] = mk16(lo, hi);
        }
        v16h bfrag[4];
#pragma unroll
        for (int j = 0; j < 4; ++j) {
            const int n = wn * 64 + j * 16 + fm;
            v8h lo = *(const v8h*)&Bs[n * LDS_STRIDE + kb];
            v8h hi = *(const v8h*)&Bs[n * LDS_STRIDE + kb + 16];
            bfrag[j] = mk16(lo, hi);
        }

        // ---- 8x v_wmma_f32_16x16x32_f16 ----
#pragma unroll
        for (int i = 0; i < 2; ++i)
#pragma unroll
            for (int j = 0; j < 4; ++j)
                acc[i][j] = __builtin_amdgcn_wmma_f32_16x16x32_f16(
                    false, afrag[i], false, bfrag[j], (short)0, acc[i][j],
                    false, false);
        __syncthreads();
    }

    // ---- epilogue: C/D layout: lane<16 -> n=lane, m=r ; lane>=16 -> n=lane-16, m=8+r
    const int nlane = lane & 15;
    const int moff  = (lane >> 4) * 8;
#pragma unroll
    for (int i = 0; i < 2; ++i) {
#pragma unroll
        for (int j = 0; j < 4; ++j) {
            const int col = n0 + wn * 64 + j * 16 + nlane;
            if (col < N) {
                const float bv = bias ? bias[col] : 0.f;
#pragma unroll
                for (int rr = 0; rr < 8; ++rr) {
                    const int row = m0 + wm * 32 + i * 16 + moff + rr;
                    float v = acc[i][j][rr] + bv;
                    if (act == ACT_GELU) {
                        v = 0.5f * v * (1.f + erff(v * 0.70710678118654752f));
                    } else if (act == ACT_SOFTPLUS) {
                        v = (v > 20.f) ? v : log1pf(__expf(v));
                    }
                    if (residual) v += residual[(size_t)row * N + col];
                    if (outF) outF[(size_t)row * N + col] = v;
                    if (outH) outH[(size_t)row * N + col] = (_Float16)v;
                }
            }
        }
    }
}

// ---------------------------------------------------------------------------
// Depthwise causal conv (len 4) + bias + SiLU. reverse=1 => anti-causal
// (equivalent to flip -> causal conv -> flip). Reads x-half of xz (f32),
// writes f16.
// ---------------------------------------------------------------------------
__global__ __launch_bounds__(256) void conv_silu_kernel(
    const float* __restrict__ xz, const float* __restrict__ cw,
    const float* __restrict__ cb, _Float16* __restrict__ out, int reverse) {
    const size_t idx = (size_t)blockIdx.x * 256 + threadIdx.x;  // b*L*DI total
    const int d = (int)(idx & 1023);
    const size_t bt = idx >> 10;
    const int t = (int)(bt & 2047);
    const size_t b = bt >> 11;
    const float* w = cw + (size_t)d * 4;   // conv_w[d][0][j]
    float acc = cb[d];
    if (!reverse) {
#pragma unroll
        for (int j = 0; j < 4; ++j) {
            const int tt = t - 3 + j;
            if (tt >= 0) acc += w[j] * xz[((b * 2048 + tt) * 2048) + d];
        }
    } else {
#pragma unroll
        for (int j = 0; j < 4; ++j) {
            const int tt = t + 3 - j;
            if (tt < 2048) acc += w[j] * xz[((b * 2048 + tt) * 2048) + d];
        }
    }
    const float s = acc / (1.f + __expf(-acc));   // SiLU
    out[idx] = (_Float16)s;
}

// ---------------------------------------------------------------------------
// Slice first 32 cols of x_dbl[8192,64] to f16 (dt_rank inputs for dt proj)
// ---------------------------------------------------------------------------
__global__ __launch_bounds__(256) void slicecvt_kernel(const float* __restrict__ xdbl,
                                                       _Float16* __restrict__ dtA, int n) {
    const int i = blockIdx.x * 256 + threadIdx.x;
    if (i < n) {
        const int r = i >> 5, c = i & 31;
        dtA[i] = (_Float16)xdbl[(size_t)r * 64 + c];
    }
}

// ---------------------------------------------------------------------------
// Selective scan. One thread per (batch, channel): 16-float state in regs,
// sequential over L (reversed for backward dir). Fuses +D*x skip and SiLU(z)
// gate; writes gated activations in f16 for the out_proj WMMA GEMM.
// ---------------------------------------------------------------------------
__global__ __launch_bounds__(256) void scan_kernel(
    const float* __restrict__ dt, const _Float16* __restrict__ xc,
    const float* __restrict__ xdbl, const float* __restrict__ xz,
    const float* __restrict__ A_log, const float* __restrict__ Dp,
    _Float16* __restrict__ yout, int reverse) {
    const int gid = blockIdx.x * 256 + threadIdx.x;   // 0..4095
    const int d = gid & 1023;
    const int b = gid >> 10;

    float Av[16];
#pragma unroll
    for (int n = 0; n < 16; ++n) Av[n] = -__expf(A_log[(size_t)d * 16 + n]);
    const float Dd = Dp[d];

    float h[16];
#pragma unroll
    for (int n = 0; n < 16; ++n) h[n] = 0.f;

    for (int s = 0; s < 2048; ++s) {
        const int t = reverse ? (2047 - s) : s;
        const size_t bt = (size_t)b * 2048 + t;
        const float dtv = dt[bt * 1024 + d];
        const float xv  = (float)xc[bt * 1024 + d];
        const float dx  = dtv * xv;
        const float* bc = xdbl + bt * 64;
        float y = 0.f;
#pragma unroll
        for (int n = 0; n < 16; ++n) {
            const float dA = __expf(dtv * Av[n]);
            h[n] = dA * h[n] + dx * bc[32 + n];
            y += h[n] * bc[48 + n];
        }
        const float z = xz[bt * 2048 + 1024 + d];
        const float g = z / (1.f + __expf(-z));     // SiLU gate
        yout[bt * 1024 + d] = (_Float16)((y + Dd * xv) * g);
    }
}

// ---------------------------------------------------------------------------
// x1 = x + 0.5*(y_fwd + y_bwd)
// ---------------------------------------------------------------------------
__global__ __launch_bounds__(256) void combine_kernel(const float* __restrict__ x,
                                                      const float* __restrict__ yf,
                                                      const float* __restrict__ yb,
                                                      float* __restrict__ x1, int n) {
    const int i = blockIdx.x * 256 + threadIdx.x;
    if (i < n) x1[i] = x[i] + 0.5f * (yf[i] + yb[i]);
}

// ---------------------------------------------------------------------------
// Host orchestration
// ---------------------------------------------------------------------------
extern "C" void kernel_launch(void* const* d_in, const int* in_sizes, int n_in,
                              void* d_out, int out_size, void* d_ws, size_t ws_size,
                              hipStream_t stream) {
    constexpr int Bsz = 4, L = 2048, DM = 512, DI = 1024, DFF = 2048;
    constexpr size_t TOK = (size_t)Bsz * L;          // 8192

    // ---- inputs (setup_inputs dict order, nested dicts flattened) ----
    const float* x    = (const float*)d_in[0];
    const float* ln_g = (const float*)d_in[1];
    const float* ln_b = (const float*)d_in[2];
    // per-direction mamba params: [in_w, conv_w, conv_b, xp_w, dt_w, dt_b, A_log, D, out_w]
    const float* P[2][9];
    for (int dir = 0; dir < 2; ++dir)
        for (int k = 0; k < 9; ++k) P[dir][k] = (const float*)d_in[3 + dir * 9 + k];
    const float* f_ln_g = (const float*)d_in[21];
    const float* f_ln_b = (const float*)d_in[22];
    const float* w1 = (const float*)d_in[23];
    const float* b1 = (const float*)d_in[24];
    const float* w2 = (const float*)d_in[25];
    const float* b2 = (const float*)d_in[26];

    // ---- workspace carve ----
    char* wsp = (char*)d_ws;
    auto carve = [&](size_t bytes) -> void* {
        void* p = (void*)wsp;
        wsp += (bytes + 255) & ~(size_t)255;
        return p;
    };
    _Float16* xn_h   = (_Float16*)carve(TOK * DM * 2);
    _Float16* inw_h[2], *xpw_h[2], *dtw_h[2], *outw_h[2];
    float*    xz_f[2]; _Float16* xconv_h[2]; float* xdbl_f[2];
    _Float16* dtA_h[2]; float* dt_f[2]; _Float16* yg_h[2]; float* ydir_f[2];
    for (int dir = 0; dir < 2; ++dir) {
        inw_h[dir]  = (_Float16*)carve((size_t)2 * DI * DM * 2);
        xpw_h[dir]  = (_Float16*)carve((size_t)64 * DI * 2);
        dtw_h[dir]  = (_Float16*)carve((size_t)DI * 32 * 2);
        outw_h[dir] = (_Float16*)carve((size_t)DM * DI * 2);
        xz_f[dir]   = (float*)carve(TOK * 2 * DI * 4);
        xconv_h[dir]= (_Float16*)carve(TOK * DI * 2);
        xdbl_f[dir] = (float*)carve(TOK * 64 * 4);
        dtA_h[dir]  = (_Float16*)carve(TOK * 32 * 2);
        dt_f[dir]   = (float*)carve(TOK * DI * 4);
        yg_h[dir]   = (_Float16*)carve(TOK * DI * 2);
        ydir_f[dir] = (float*)carve(TOK * DM * 4);
    }
    float*    x1_f = (float*)carve(TOK * DM * 4);
    _Float16* hf_h = (_Float16*)carve(TOK * DM * 2);
    _Float16* gf_h = (_Float16*)carve(TOK * DFF * 2);
    _Float16* w1_h = (_Float16*)carve((size_t)DFF * DM * 2);
    _Float16* w2_h = (_Float16*)carve((size_t)DM * DFF * 2);

    auto cvt = [&](const float* src, _Float16* dst, int n) {
        cvt_f32_f16<<<(n + 255) / 256, 256, 0, stream>>>(src, dst, n);
    };

    // ---- weight downconversion to f16 ----
    for (int dir = 0; dir < 2; ++dir) {
        cvt(P[dir][0], inw_h[dir],  2 * DI * DM);
        cvt(P[dir][3], xpw_h[dir],  64 * DI);
        cvt(P[dir][4], dtw_h[dir],  DI * 32);
        cvt(P[dir][8], outw_h[dir], DM * DI);
    }
    cvt(w1, w1_h, DFF * DM);
    cvt(w2, w2_h, DM * DFF);

    // ---- LN1 ----
    ln_kernel<<<(int)TOK, 256, 0, stream>>>(x, ln_g, ln_b, xn_h, DM);

    // ---- per-direction mamba ----
    for (int dir = 0; dir < 2; ++dir) {
        // xz = xn @ in_w^T                       [8192, 2048]
        gemm_wmma_f16<<<dim3((2 * DI) / TILE_N, TOK / TILE_M), 256, 0, stream>>>(
            xn_h, inw_h[dir], nullptr, nullptr, xz_f[dir], nullptr,
            (int)TOK, 2 * DI, DM, ACT_NONE);
        // depthwise conv + SiLU -> xconv (f16)
        conv_silu_kernel<<<(int)(TOK * DI / 256), 256, 0, stream>>>(
            xz_f[dir], P[dir][1], P[dir][2], xconv_h[dir], dir);
        // x_dbl = xconv @ xp_w^T                 [8192, 64]
        gemm_wmma_f16<<<dim3(1, TOK / TILE_M), 256, 0, stream>>>(
            xconv_h[dir], xpw_h[dir], nullptr, nullptr, xdbl_f[dir], nullptr,
            (int)TOK, 64, DI, ACT_NONE);
        // dt input slice -> f16
        slicecvt_kernel<<<(int)(TOK * 32 / 256), 256, 0, stream>>>(
            xdbl_f[dir], dtA_h[dir], (int)(TOK * 32));
        // dt = softplus(dtA @ dt_w^T + dt_b)     [8192, 1024]
        gemm_wmma_f16<<<dim3(DI / TILE_N, TOK / TILE_M), 256, 0, stream>>>(
            dtA_h[dir], dtw_h[dir], P[dir][5], nullptr, dt_f[dir], nullptr,
            (int)TOK, DI, 32, ACT_SOFTPLUS);
        // selective scan (+D*x skip, SiLU(z) gate) -> yg (f16)
        scan_kernel<<<16, 256, 0, stream>>>(
            dt_f[dir], xconv_h[dir], xdbl_f[dir], xz_f[dir],
            P[dir][6], P[dir][7], yg_h[dir], dir);
        // y = yg @ out_w^T                       [8192, 512]
        gemm_wmma_f16<<<dim3(DM / TILE_N, TOK / TILE_M), 256, 0, stream>>>(
            yg_h[dir], outw_h[dir], nullptr, nullptr, ydir_f[dir], nullptr,
            (int)TOK, DM, DI, ACT_NONE);
    }

    // ---- bidirectional mean fusion + residual ----
    combine_kernel<<<(int)(TOK * DM / 256), 256, 0, stream>>>(
        x, ydir_f[0], ydir_f[1], x1_f, (int)(TOK * DM));

    // ---- FFN ----
    ln_kernel<<<(int)TOK, 256, 0, stream>>>(x1_f, f_ln_g, f_ln_b, hf_h, DM);
    // g = gelu(h @ w1^T + b1)                    [8192, 2048] (f16)
    gemm_wmma_f16<<<dim3(DFF / TILE_N, TOK / TILE_M), 256, 0, stream>>>(
        hf_h, w1_h, b1, nullptr, nullptr, gf_h, (int)TOK, DFF, DM, ACT_GELU);
    // out = x1 + g @ w2^T + b2                   [8192, 512] (f32 -> d_out)
    gemm_wmma_f16<<<dim3(DM / TILE_N, TOK / TILE_M), 256, 0, stream>>>(
        gf_h, w2_h, b2, x1_f, (float*)d_out, nullptr, (int)TOK, DM, DFF, ACT_NONE);
}
